// CIN_2637109920285
// MI455X (gfx1250) — matrix operand order repaired and checked by reading, not compiled, not run on hardware
//
#include <hip/hip_runtime.h>
#include <hip/hip_bf16.h>

typedef __bf16 v16bf __attribute__((ext_vector_type(16)));
typedef __bf16 v8bf  __attribute__((ext_vector_type(8)));
typedef __bf16 v2bf  __attribute__((ext_vector_type(2)));
typedef float  v8f   __attribute__((ext_vector_type(8)));

#define B_    1024
#define F_    39
#define D_    32
#define CL_   400
#define CLP_  416         // W rows padded to 13 pairs of 16
#define HALF_ 200
#define N_    (B_ * D_)   // 32768 columns (b,d)
#define GP1   64          // padded G for layer 1 (G=39)
#define GP2   224         // padded G for layers 2,3 (G=200)
#define POOLC 800
#define OUT_  400
#define XSPW  40          // dwords per column in pre-splatted x buffer

#if __has_builtin(__builtin_amdgcn_sched_group_barrier)
#define SGB(mask, size, id) __builtin_amdgcn_sched_group_barrier(mask, size, id)
#else
#define SGB(mask, size, id)
#endif

// ---------------- prep kernels ----------------

__global__ void k_zero_f32(float* p, int n) {
    int i = blockIdx.x * blockDim.x + threadIdx.x;
    if (i < n) p[i] = 0.0f;
}

// x (B,F,D) fp32 -> xT [n][GP1] bf16 (H operand for layer 1, pad rows zero)
//               -> xSp [n][40] dwords, each = bf16 value duplicated in both halves
__global__ void k_build_x(const float* __restrict__ x, __bf16* __restrict__ xT,
                          unsigned int* __restrict__ xSp) {
    int n = blockIdx.x * blockDim.x + threadIdx.x;
    if (n >= N_) return;
    int b = n >> 5, d = n & 31;
    __bf16* r = xT + (size_t)n * GP1;
    unsigned int* s = xSp + (size_t)n * XSPW;
    const float* xb = x + (size_t)b * (F_ * D_) + d;
#pragma unroll
    for (int f = 0; f < F_; ++f) {
        __bf16 v = (__bf16)xb[f * D_];
        r[f] = v;
        unsigned short u = __builtin_bit_cast(unsigned short, v);
        s[f] = ((unsigned int)u << 16) | u;
    }
#pragma unroll
    for (int f = F_; f < GP1; ++f) r[f] = (__bf16)0.0f;
    s[F_] = 0u;
}

// zero the pad rows (g in [200,224)) of both h buffers
__global__ void k_zero_hpad(__bf16* __restrict__ h1, __bf16* __restrict__ h2) {
    int n = blockIdx.x * blockDim.x + threadIdx.x;
    if (n >= N_) return;
    for (int g = HALF_; g < GP2; ++g) {
        h1[(size_t)n * GP2 + g] = (__bf16)0.0f;
        h2[(size_t)n * GP2 + g] = (__bf16)0.0f;
    }
}

// w (400,F,Gsrc) fp32 -> (416,F,Gpad) bf16, zero padded in g and in o
__global__ void k_conv_w(const float* __restrict__ src, __bf16* __restrict__ dst,
                         int Gsrc, int Gpad, int total) {
    int i = blockIdx.x * blockDim.x + threadIdx.x;
    if (i >= total) return;
    int g  = i % Gpad;
    int of = i / Gpad;           // o * F_ + f
    int o  = of / F_;
    dst[i] = (g < Gsrc && o < CL_) ? (__bf16)src[(size_t)of * Gsrc + g] : (__bf16)0.0f;
}

// ---------------- CIN layer: Z = W * (x ⊗ h) via v_wmma_f32_16x16x32_bf16 ----------------
// Each wave computes a 32(O) x 16(N) tile (two 16x16 WMMA tiles sharing one B fragment).
// K = F_*GPAD tiled as (f, 32-wide g chunk); within a chunk f is constant so the B fragment
// is splat(x0[f,n]) * H-fragment -> 8 v_pk_mul_bf16, amortized over 2 WMMAs.
// Depth-2 rotation for BOTH the A fragments and the bv fragments, with a
// sched_group_barrier pattern {2 VMEM, 4 VALU, 1 WMMA}x2 per f to pin the software
// pipeline (keeps ~8 loads in flight, and puts 8 VALU between a WMMA and the next
// overwrite of its operands so no hazard v_nops are needed).

__device__ __forceinline__ v16bf mulbc(v16bf h, unsigned int s) {
    union { unsigned int u; v2bf p; } c; c.u = s;
    union { v16bf v; v2bf p[8]; } a, r;
    a.v = h;
#pragma unroll
    for (int i = 0; i < 8; ++i) r.p[i] = a.p[i] * c.p;
    return r.v;
}

template<int GPAD>
__device__ __forceinline__ v16bf ldA(const __bf16* wc, int f) {
    v8bf lo = *(const v8bf*)(wc + f * GPAD);
    v8bf hi = *(const v8bf*)(wc + f * GPAD + 8);
    return __builtin_shufflevector(lo, hi,
        0, 1, 2, 3, 4, 5, 6, 7, 8, 9, 10, 11, 12, 13, 14, 15);
}

template<int GPAD, int NGC, bool WRITE_H>
__global__ __launch_bounds__(128, 4) void k_cin(
    const __bf16* __restrict__ wb,        // [416][39][GPAD] bf16 (zero padded)
    const __bf16* __restrict__ hin,       // [N][GPAD] bf16
    const unsigned int* __restrict__ xSp, // [N][40] pre-splatted x
    const float*  __restrict__ bias,      // [400]
    __bf16* __restrict__ hout,            // [N][GP2] (unused when !WRITE_H)
    float* __restrict__ pooled,           // [B][800]
    int colbase)
{
    const int lane = threadIdx.x & 31;
    const int wave = threadIdx.x >> 5;                 // 0..3
    const int pair = blockIdx.y * 4 + wave;            // 0..15
    if (pair >= 13) return;                            // 13 pairs cover 416 padded rows
    const int n0 = blockIdx.x * 16;                    // 16 columns = 16 d's of one batch
    const int nl = lane & 15;
    const int kh = lane >> 4;
    const int ob = pair * 32;

    // per-lane pre-splatted x0[f, n] (dword = bf16 duplicated)
    unsigned int xsp[XSPW];
    {
        const uint4* xr = (const uint4*)(xSp + (size_t)(n0 + nl) * XSPW);
        uint4* xd = (uint4*)xsp;
#pragma unroll
        for (int i = 0; i < XSPW / 4; ++i) xd[i] = xr[i];
    }

    // B fragment source: lane half kh holds K = kh*16 .. kh*16+15
    const __bf16* hrow = hin + (size_t)(n0 + nl) * GPAD + kh * 16;
    // A fragment sources for the two M-tiles (contiguous K layout matching B: kh -> K kh*16 .. kh*16+15)
    const __bf16* wrow0 = wb + (size_t)(ob      + nl) * (F_ * GPAD) + kh * 16;
    const __bf16* wrow1 = wb + (size_t)(ob + 16 + nl) * (F_ * GPAD) + kh * 16;

    v8f acc0 = {}, acc1 = {};
    for (int gc = 0; gc < NGC; ++gc) {
        v16bf hv = *(const v16bf*)(hrow + gc * 32);
        const __bf16* w0c = wrow0 + gc * 32;
        const __bf16* w1c = wrow1 + gc * 32;

        v16bf a0[3], a1[3], bvp[3];          // depth-2 rotating pipeline slots
        a0[0] = ldA<GPAD>(w0c, 0); a1[0] = ldA<GPAD>(w1c, 0);
        a0[1] = ldA<GPAD>(w0c, 1); a1[1] = ldA<GPAD>(w1c, 1);
        bvp[0] = mulbc(hv, xsp[0]);
        bvp[1] = mulbc(hv, xsp[1]);
#pragma unroll
        for (int f = 0; f < F_; ++f) {
            if (f + 2 < F_) {
                a0[(f + 2) % 3] = ldA<GPAD>(w0c, f + 2);
                a1[(f + 2) % 3] = ldA<GPAD>(w1c, f + 2);
                bvp[(f + 2) % 3] = mulbc(hv, xsp[f + 2]);
            }
            acc0 = __builtin_amdgcn_wmma_f32_16x16x32_bf16(
                false, a0[f % 3], false, bvp[f % 3], (short)0, acc0, false, false);
            acc1 = __builtin_amdgcn_wmma_f32_16x16x32_bf16(
                false, a1[f % 3], false, bvp[f % 3], (short)0, acc1, false, false);
            // steady-state pipeline pattern: {2 VMEM-read, 4 VALU, 1 WMMA} x 2
            SGB(0x020, 2, 0);   // vmem reads
            SGB(0x002, 4, 0);   // valu (pk_mul_bf16)
            SGB(0x008, 1, 0);   // wmma
            SGB(0x020, 2, 0);
            SGB(0x002, 4, 0);
            SGB(0x008, 1, 0);
        }
    }

    // epilogue: bias + relu; C layout: VGPR v -> M = v + 8*kh, N = lane%16
    const int b = n0 >> 5;
    float zr[16], rs[16];
#pragma unroll
    for (int t = 0; t < 2; ++t) {
        const v8f& acc = t ? acc1 : acc0;
#pragma unroll
        for (int v = 0; v < 8; ++v) {
            int o = ob + t * 16 + v + 8 * kh;
            float z = acc[v] + (o < CL_ ? bias[o] : 0.0f);
            zr[t * 8 + v] = z > 0.0f ? z : 0.0f;
        }
    }
    // reduce over the 16 columns (d) for pooled sum(axis=d)
#pragma unroll
    for (int i = 0; i < 16; ++i) {
        float s = zr[i];
        s += __shfl_xor(s, 1, 32);
        s += __shfl_xor(s, 2, 32);
        s += __shfl_xor(s, 4, 32);
        s += __shfl_xor(s, 8, 32);
        rs[i] = s;
    }
#pragma unroll
    for (int t = 0; t < 2; ++t) {
#pragma unroll
        for (int v = 0; v < 8; ++v) {
            int o = ob + t * 16 + v + 8 * kh;
            if (o >= CL_) continue;                       // padded W rows
            bool toH = WRITE_H && (o >= HALF_);
            if (toH) {
                hout[(size_t)(n0 + nl) * GP2 + (o - HALF_)] = (__bf16)zr[t * 8 + v];
            } else if (nl == 0) {
                atomicAdd(&pooled[(size_t)b * POOLC + colbase + o], rs[t * 8 + v]);
            }
        }
    }
}

// ---------------- final projection + batchnorm ----------------

__global__ void k_outmm(const float* __restrict__ pooled, const float* __restrict__ ow,
                        const float* __restrict__ ob, float* __restrict__ out) {
    int i = blockIdx.x * blockDim.x + threadIdx.x;
    if (i >= B_ * OUT_) return;
    int b = i / OUT_, j = i % OUT_;
    const float* pr = pooled + (size_t)b * POOLC;
    float s = ob[j];
    for (int k = 0; k < POOLC; ++k) s = fmaf(pr[k], ow[(size_t)k * OUT_ + j], s);
    out[i] = s;
}

__global__ __launch_bounds__(256) void k_bn(const float* __restrict__ out,
                                            const float* __restrict__ gamma,
                                            const float* __restrict__ beta,
                                            float* __restrict__ y) {
    __shared__ float s1[256], s2[256];
    int j = blockIdx.x, t = threadIdx.x;
    float a = 0.0f, b = 0.0f;
    for (int r = t; r < B_; r += 256) {
        float v = out[(size_t)r * OUT_ + j];
        a += v; b += v * v;
    }
    s1[t] = a; s2[t] = b;
    __syncthreads();
    for (int st = 128; st > 0; st >>= 1) {
        if (t < st) { s1[t] += s1[t + st]; s2[t] += s2[t + st]; }
        __syncthreads();
    }
    float mean = s1[0] * (1.0f / B_);
    float var  = s2[0] * (1.0f / B_) - mean * mean;
    float inv  = rsqrtf(var + 1e-5f) * gamma[j];
    float bb   = beta[j];
    for (int r = t; r < B_; r += 256) {
        float v = (out[(size_t)r * OUT_ + j] - mean) * inv + bb;
        y[(size_t)r * OUT_ + j] = v > 0.0f ? v : 0.0f;
    }
}

// ---------------- launch ----------------

extern "C" void kernel_launch(void* const* d_in, const int* in_sizes, int n_in,
                              void* d_out, int out_size, void* d_ws, size_t ws_size,
                              hipStream_t stream) {
    const float* x   = (const float*)d_in[0];
    const float* w1  = (const float*)d_in[1];
    const float* b1  = (const float*)d_in[2];
    const float* w2  = (const float*)d_in[3];
    const float* b2  = (const float*)d_in[4];
    const float* w3  = (const float*)d_in[5];
    const float* b3  = (const float*)d_in[6];
    const float* ow  = (const float*)d_in[7];
    const float* obi = (const float*)d_in[8];
    const float* gamma = (const float*)d_in[9];
    const float* beta  = (const float*)d_in[10];

    char* ws = (char*)d_ws;
    size_t off = 0;
    auto alloc = [&](size_t bytes) -> char* {
        char* p = ws + off;
        off += (bytes + 255) & ~(size_t)255;
        return p;
    };
    float*  pooled = (float*) alloc((size_t)B_ * POOLC * 4);
    float*  outbuf = (float*) alloc((size_t)B_ * OUT_ * 4);
    __bf16* xT     = (__bf16*)alloc((size_t)N_ * GP1 * 2);
    unsigned int* xSp = (unsigned int*)alloc((size_t)N_ * XSPW * 4);
    __bf16* h1     = (__bf16*)alloc((size_t)N_ * GP2 * 2);
    __bf16* h2     = (__bf16*)alloc((size_t)N_ * GP2 * 2);
    __bf16* w1b    = (__bf16*)alloc((size_t)CLP_ * F_ * GP1 * 2);
    __bf16* w2b    = (__bf16*)alloc((size_t)CLP_ * F_ * GP2 * 2);
    __bf16* w3b    = (__bf16*)alloc((size_t)CLP_ * F_ * GP2 * 2);

    // prep (deterministic each call; no state carried across calls)
    k_zero_f32<<<(B_ * POOLC + 255) / 256, 256, 0, stream>>>(pooled, B_ * POOLC);
    k_build_x<<<(N_ + 255) / 256, 256, 0, stream>>>(x, xT, xSp);
    k_zero_hpad<<<(N_ + 255) / 256, 256, 0, stream>>>(h1, h2);
    k_conv_w<<<(CLP_ * F_ * GP1 + 255) / 256, 256, 0, stream>>>(w1, w1b, F_,    GP1, CLP_ * F_ * GP1);
    k_conv_w<<<(CLP_ * F_ * GP2 + 255) / 256, 256, 0, stream>>>(w2, w2b, HALF_, GP2, CLP_ * F_ * GP2);
    k_conv_w<<<(CLP_ * F_ * GP2 + 255) / 256, 256, 0, stream>>>(w3, w3b, HALF_, GP2, CLP_ * F_ * GP2);

    // 3 CIN layers: grid.x = N-tiles (2048), grid.y*4 waves = 16 pair-slots (13 used)
    dim3 grid(N_ / 16, 4), blk(128);
    k_cin<GP1, GP1 / 32, true ><<<grid, blk, 0, stream>>>(w1b, xT, xSp, b1, h1, pooled, 0);
    k_cin<GP2, GP2 / 32, true ><<<grid, blk, 0, stream>>>(w2b, h1, xSp, b2, h2, pooled, 200);
    k_cin<GP2, GP2 / 32, false><<<grid, blk, 0, stream>>>(w3b, h2, xSp, b3, nullptr, pooled, 400);

    // pooled @ out_w + out_b, then batchnorm + relu
    k_outmm<<<(B_ * OUT_ + 255) / 256, 256, 0, stream>>>(pooled, ow, obi, outbuf);
    k_bn<<<OUT_, 256, 0, stream>>>(outbuf, gamma, beta, (float*)d_out);
}
